// CachedMixtralAttention_54752243089676
// MI455X (gfx1250) — matrix operand order repaired and checked by reading, not compiled, not run on hardware
//
#include <hip/hip_runtime.h>
#include <hip/hip_bf16.h>

// ---------------------------------------------------------------------------
// Types for CDNA5 WMMA (wave32): v_wmma_f32_16x16x32_bf16
// ---------------------------------------------------------------------------
typedef __attribute__((ext_vector_type(16))) __bf16 v16bf;
typedef __attribute__((ext_vector_type(8)))  float  v8f;
typedef __attribute__((ext_vector_type(4)))  unsigned int v4u;
typedef __attribute__((ext_vector_type(8)))  int v8i;
typedef __attribute__((ext_vector_type(4)))  int v4i;

union FragB { unsigned int u[8]; v16bf v; };

__device__ __forceinline__ unsigned short f2bf(float f) {
  unsigned int u = __float_as_uint(f);
  u += 0x7FFFu + ((u >> 16) & 1u);          // round-to-nearest-even
  return (unsigned short)(u >> 16);
}
__device__ __forceinline__ unsigned int pk2(float a, float b) {
  return (unsigned int)f2bf(a) | ((unsigned int)f2bf(b) << 16);
}

// ---------------------------------------------------------------------------
// Generic GEMM: C[M,N] = A[M,K] * B[K,N], fp32 in/out, bf16 WMMA inside.
// Block tile 128x128, K-step 64. 256 threads = 8 waves tiled 4(M) x 2(N);
// each wave owns a 32x64 C sub-tile (2 A-frags reuse 4 B-frags -> 8 WMMA).
// Software pipeline: global float4 loads for tile k+1 overlap compute of k.
// ---------------------------------------------------------------------------
#define BM 128
#define BN 128
#define BK 64
#define ASTR 66   // halves per row; 66*2/4 = 33 bank step, coprime with 64
#define BSTR 66

__global__ __launch_bounds__(256) void gemm_f32_bf16wmma(
    const float* __restrict__ A, const float* __restrict__ B,
    float* __restrict__ C, int M, int N, int K)
{
  __shared__ alignas(16) unsigned short As[BM * ASTR];   // [m][k]
  __shared__ alignas(16) unsigned short Bs[BN * BSTR];   // transposed: [n][k]
  const int t = threadIdx.x;
  const int w = t >> 5, lane = t & 31;
  const int wm = w & 3, wn = w >> 2;
  const int m = lane & 15, half = lane >> 4;
  const int m0 = blockIdx.y * BM;
  const int n0 = blockIdx.x * BN;

  // A tile staging map: 128 rows x 16 float4 (64 cols)
  const int ar_ = t >> 4, ac_ = (t & 15) << 2;     // +32 rows per step
  // B tile staging map: 64 rows x 32 float4 (128 cols)
  const int br_ = t >> 5, bc_ = (t & 31) << 2;     // +8 rows per step

  v8f acc[2][4];
#pragma unroll
  for (int sm = 0; sm < 2; ++sm)
#pragma unroll
    for (int c = 0; c < 4; ++c)
      acc[sm][c] = (v8f){0.f,0.f,0.f,0.f,0.f,0.f,0.f,0.f};

  float4 ra[8], rb[8];
  // prologue: fetch tile kt=0 into registers
#pragma unroll
  for (int i = 0; i < 8; ++i)
    ra[i] = *(const float4*)(A + (size_t)(m0 + ar_ + 32 * i) * K + ac_);
#pragma unroll
  for (int i = 0; i < 8; ++i)
    rb[i] = *(const float4*)(B + (size_t)(br_ + 8 * i) * N + n0 + bc_);

  for (int kt = 0; kt < K; kt += BK) {
    // commit current registers to LDS (convert fp32 -> bf16)
#pragma unroll
    for (int i = 0; i < 8; ++i) {
      int row = ar_ + 32 * i;
      *(unsigned int*)&As[row * ASTR + ac_]     = pk2(ra[i].x, ra[i].y);
      *(unsigned int*)&As[row * ASTR + ac_ + 2] = pk2(ra[i].z, ra[i].w);
    }
#pragma unroll
    for (int i = 0; i < 8; ++i) {
      int row = br_ + 8 * i;
      Bs[(bc_ + 0) * BSTR + row] = f2bf(rb[i].x);
      Bs[(bc_ + 1) * BSTR + row] = f2bf(rb[i].y);
      Bs[(bc_ + 2) * BSTR + row] = f2bf(rb[i].z);
      Bs[(bc_ + 3) * BSTR + row] = f2bf(rb[i].w);
    }
    __syncthreads();

    // issue next tile's global loads; they overlap the WMMA compute below
    if (kt + BK < K) {
      const float* An = A + kt + BK;
      const float* Bn = B + (size_t)(kt + BK) * N;
#pragma unroll
      for (int i = 0; i < 8; ++i)
        ra[i] = *(const float4*)(An + (size_t)(m0 + ar_ + 32 * i) * K + ac_);
#pragma unroll
      for (int i = 0; i < 8; ++i)
        rb[i] = *(const float4*)(Bn + (size_t)(br_ + 8 * i) * N + n0 + bc_);
    }

#pragma unroll
    for (int kc = 0; kc < BK / 32; ++kc) {
      // two A fragments (rows 32*wm and 32*wm+16), ISA A-layout
      FragB af[2];
#pragma unroll
      for (int sm = 0; sm < 2; ++sm) {
        const unsigned short* arp = &As[(32 * wm + 16 * sm + m) * ASTR + kc * 32];
#pragma unroll
        for (int v = 0; v < 8; ++v) {
          int kk = ((v & 4) ? 16 : 0) + 8 * half + (v & 3) * 2;
          af[sm].u[v] = *(const unsigned int*)(arp + kk);
        }
      }
#pragma unroll
      for (int c = 0; c < 4; ++c) {
        // B fragment: lane n = 64*wn + c*16 + m, K = ((v&4)?16:0) + 8*half + 2*(v&3)
        FragB bf_;
        const unsigned short* brp =
            &Bs[(64 * wn + c * 16 + m) * BSTR + kc * 32];
#pragma unroll
        for (int v = 0; v < 8; ++v)
          bf_.u[v] = *(const unsigned int*)(brp + ((v & 4) ? 16 : 0) + 8 * half + (v & 3) * 2);
        acc[0][c] = __builtin_amdgcn_wmma_f32_16x16x32_bf16(
            false, af[0].v, false, bf_.v, (short)0, acc[0][c], false, false);
        acc[1][c] = __builtin_amdgcn_wmma_f32_16x16x32_bf16(
            false, af[1].v, false, bf_.v, (short)0, acc[1][c], false, false);
      }
    }
    __syncthreads();
  }

  // epilogue: C layout M = r + 8*half, N = lane&15
#pragma unroll
  for (int sm = 0; sm < 2; ++sm)
#pragma unroll
    for (int c = 0; c < 4; ++c)
#pragma unroll
      for (int r = 0; r < 8; ++r)
        C[(size_t)(m0 + 32 * wm + 16 * sm + r + 8 * half) * N +
          n0 + 64 * wn + c * 16 + m] = acc[sm][c][r];
}

// ---------------------------------------------------------------------------
// RoPE (rotate-half) + transpose [s][H][128] fp32 -> [H][s][128] bf16
// ---------------------------------------------------------------------------
__global__ void rope_transpose_bf16(const float* __restrict__ src,
                                    unsigned short* __restrict__ dst, int H)
{
  int t = blockIdx.x * blockDim.x + threadIdx.x;
  int total = 2048 * H * 64;
  if (t >= total) return;
  int i = t & 63;
  int h = (t >> 6) % H;
  int s = t / (64 * H);
  const float* xp = src + ((size_t)s * H + h) * 128;
  float x1 = xp[i], x2 = xp[i + 64];
  float inv_freq = __expf(-0.1439115660453411f * (float)i);  // ln(10000)/64
  float ang = (float)s * inv_freq;
  float sn, cs;
  sincosf(ang, &sn, &cs);
  unsigned short* op = dst + ((size_t)h * 2048 + s) * 128;
  op[i]      = f2bf(x1 * cs - x2 * sn);
  op[i + 64] = f2bf(x2 * cs + x1 * sn);
}

// ---------------------------------------------------------------------------
// V convert + transpose [s][8][128] fp32 -> [8][s][128] bf16
// ---------------------------------------------------------------------------
__global__ void conv_transpose_bf16(const float* __restrict__ src,
                                    unsigned short* __restrict__ dst, int H)
{
  int t = blockIdx.x * blockDim.x + threadIdx.x;
  int total = 2048 * H * 128;
  if (t >= total) return;
  int d = t & 127;
  int h = (t >> 7) % H;
  int s = t / (128 * H);
  dst[((size_t)h * 2048 + s) * 128 + d] = f2bf(src[((size_t)s * H + h) * 128 + d]);
}

// ---------------------------------------------------------------------------
// Sliding-window flash attention. Block = 1 head x 64 q-rows (4 waves x 16).
// Key loop restricted to window [q0-511, q0+63] -> <=576 of 2048 keys (~3.6x less score work).
// K tile fetched by the Tensor Data Mover (TENSOR_LOAD_TO_LDS) with hardware
// row padding (1 DWORD per 64-DWORD row -> KTSTR=130 bank-conflict-free);
// V tile loaded/transposed manually (TDM cannot transpose 16-bit data).
// ---------------------------------------------------------------------------
#define KTSTR 130
#define VTSTR 34
#define PSTR  34

__global__ __launch_bounds__(128) void attn_swa_kernel(
    const unsigned short* __restrict__ qb,
    const unsigned short* __restrict__ kb,
    const unsigned short* __restrict__ vb,
    float* __restrict__ ob)
{
  __shared__ alignas(16) unsigned short Kt[32 * KTSTR];   // K tile, row-major [key][dim]
  __shared__ alignas(16) unsigned short Vt[128 * VTSTR];  // V tile transposed [dim][key]
  __shared__ alignas(16) unsigned short Pb[4 * 16 * PSTR];// per-wave P re-layout buffer

  const int t = threadIdx.x;
  const int w = t >> 5, lane = t & 31;
  const int m = lane & 15, half = lane >> 4;
  const int u = blockIdx.x;      // q head
  const int kvh = u >> 2;        // GQA: 4 q-heads per kv-head
  const int q0 = blockIdx.y * 64;
  const int qt = q0 + 16 * w;

  // preload q fragments for this wave's 16x128 tile (A layout)
  FragB qf[4];
  const unsigned short* qrow = qb + ((size_t)u * 2048 + qt + m) * 128;
#pragma unroll
  for (int kc = 0; kc < 4; ++kc)
#pragma unroll
    for (int v = 0; v < 8; ++v) {
      int kk = kc * 32 + ((v & 4) ? 16 : 0) + 8 * half + (v & 3) * 2;
      qf[kc].u[v] = *(const unsigned int*)(qrow + kk);
    }

  v8f accO[8];
#pragma unroll
  for (int j = 0; j < 8; ++j) accO[j] = (v8f){0.f,0.f,0.f,0.f,0.f,0.f,0.f,0.f};
  float mrow[8], lrow[8];
#pragma unroll
  for (int r = 0; r < 8; ++r) { mrow[r] = -1e9f; lrow[r] = 0.f; }

  int kbeg = q0 - 511; if (kbeg < 0) kbeg = 0; kbeg &= ~31;
  unsigned short* Pw = &Pb[w * 16 * PSTR];

  for (int k0 = kbeg; k0 < q0 + 64; k0 += 32) {
    __syncthreads();   // previous chunk's compute done; LDS tiles reusable

    // --- K tile via Tensor Data Mover (issued once, by wave 0) ---
    if (w == 0) {
      unsigned long long ga =
          (unsigned long long)(const void*)(kb + ((size_t)kvh * 2048 + k0) * 128);
      v4u g0;
      g0[0] = 1u;                                    // count=1, user descriptor
      g0[1] = (unsigned int)(unsigned long long)(size_t)&Kt[0];  // lds_addr
      g0[2] = (unsigned int)(ga & 0xFFFFFFFFull);    // global_addr[31:0]
      g0[3] = (unsigned int)((ga >> 32) & 0x1FFFFFFull) | (2u << 30); // [56:32] | type=2
      v8i g1;
      g1[0] = (int)((1u << 16)     // data_size = 2 bytes
                  | (1u << 20)     // pad_enable
                  | (5u << 22)     // pad_interval: 2^(5+1)=64 DWORDs (=one 128-elem row)
                  | (0u << 25));   // pad_amount: 1 DWORD  -> row stride 130 halves
      g1[1] = (int)(128u << 16);   // tensor_dim0 = 128 (low 16 bits)
      g1[2] = (int)(2048u << 16);  // tensor_dim1 = 2048 (low 16 bits)
      g1[3] = (int)(128u << 16);   // tile_dim0 = 128
      g1[4] = 32;                  // tile_dim1 = 32, tile_dim2 = 0
      g1[5] = 128;                 // tensor_dim0_stride = 128 (low 32)
      g1[6] = 0;                   // stride high bits / dim1_stride = 0
      g1[7] = 0;
      v4i gz4 = (v4i){0, 0, 0, 0};              // <=2-D tensor: groups 2/3 unused
      v8i gz8 = (v8i){0, 0, 0, 0, 0, 0, 0, 0};  // extra group (6-arg toolchain form)
      __builtin_amdgcn_tensor_load_to_lds(g0, g1, gz4, gz4, gz8, 0);
    }

    // --- V tile: manual transposed load (vector path) ---
    const unsigned short* vg = vb + ((size_t)kvh * 2048 + k0) * 128;
#pragma unroll
    for (int i = 0; i < 16; ++i) {
      int e = t + i * 128;
      int row = e >> 6, c2 = (e & 63) * 2;
      unsigned int vv = *(const unsigned int*)(vg + (size_t)row * 128 + c2);
      Vt[(c2 + 0) * VTSTR + row] = (unsigned short)(vv & 0xFFFFu);
      Vt[(c2 + 1) * VTSTR + row] = (unsigned short)(vv >> 16);
    }
    if (w == 0) __builtin_amdgcn_s_wait_tensorcnt(0);
    __syncthreads();   // publish Kt (TDM) + Vt (vector) to all waves

    // scores S = q * k^T for 32 keys (two 16x16 tiles), K-dim 128 in 4 chunks
    v8f S0 = (v8f){0.f,0.f,0.f,0.f,0.f,0.f,0.f,0.f};
    v8f S1 = (v8f){0.f,0.f,0.f,0.f,0.f,0.f,0.f,0.f};
#pragma unroll
    for (int kc = 0; kc < 4; ++kc) {
      FragB b0, b1;
      int base = kc * 32;
#pragma unroll
      for (int v = 0; v < 8; ++v) {
        b0.u[v] = *(const unsigned int*)&Kt[m * KTSTR + base + ((v & 4) ? 16 : 0) + 8 * half + (v & 3) * 2];
        b1.u[v] = *(const unsigned int*)&Kt[(m + 16) * KTSTR + base + ((v & 4) ? 16 : 0) + 8 * half + (v & 3) * 2];
      }
      S0 = __builtin_amdgcn_wmma_f32_16x16x32_bf16(false, qf[kc].v, false, b0.v, (short)0, S0, false, false);
      S1 = __builtin_amdgcn_wmma_f32_16x16x32_bf16(false, qf[kc].v, false, b1.v, (short)0, S1, false, false);
    }

    // masked online softmax (fp32). Row M = r + 8*half, key N = lane&15.
    float p0[8], p1[8], scl[8];
#pragma unroll
    for (int r = 0; r < 8; ++r) {
      int Q = qt + r + 8 * half;
      int d0 = Q - (k0 + m), d1 = Q - (k0 + 16 + m);
      float s0 = (d0 >= 0 && d0 < 512) ? S0[r] * 0.08838834764831845f : -1e9f;
      float s1 = (d1 >= 0 && d1 < 512) ? S1[r] * 0.08838834764831845f : -1e9f;
      float mx = fmaxf(s0, s1);
      mx = fmaxf(mx, __shfl_xor(mx, 1, 16));
      mx = fmaxf(mx, __shfl_xor(mx, 2, 16));
      mx = fmaxf(mx, __shfl_xor(mx, 4, 16));
      mx = fmaxf(mx, __shfl_xor(mx, 8, 16));
      float mn = fmaxf(mrow[r], mx);
      float sc = __expf(mrow[r] - mn);
      mrow[r] = mn;
      float e0 = __expf(s0 - mn), e1 = __expf(s1 - mn);
      float rs = e0 + e1;
      rs += __shfl_xor(rs, 1, 16);
      rs += __shfl_xor(rs, 2, 16);
      rs += __shfl_xor(rs, 4, 16);
      rs += __shfl_xor(rs, 8, 16);
      lrow[r] = lrow[r] * sc + rs;
      p0[r] = e0; p1[r] = e1; scl[r] = sc;
    }
#pragma unroll
    for (int j = 0; j < 8; ++j)
#pragma unroll
      for (int r = 0; r < 8; ++r) accO[j][r] *= scl[r];

    // re-layout P (C-layout -> A-fragment layout) through LDS
#pragma unroll
    for (int r = 0; r < 8; ++r) {
      int row = r + 8 * half;
      Pw[row * PSTR + m]      = f2bf(p0[r]);
      Pw[row * PSTR + m + 16] = f2bf(p1[r]);
    }
    asm volatile("s_wait_dscnt 0x0" ::: "memory");  // DS ops are wave-in-order; be explicit
    FragB pf;
#pragma unroll
    for (int v = 0; v < 8; ++v) {
      int kk = ((v & 4) ? 16 : 0) + 8 * half + (v & 3) * 2;
      pf.u[v] = *(const unsigned int*)&Pw[m * PSTR + kk];
    }
    // O += P * V over 8 head-dim chunks
#pragma unroll
    for (int j = 0; j < 8; ++j) {
      FragB vf;
#pragma unroll
      for (int v = 0; v < 8; ++v)
        vf.u[v] = *(const unsigned int*)&Vt[(j * 16 + m) * VTSTR + ((v & 4) ? 16 : 0) + 8 * half + (v & 3) * 2];
      accO[j] = __builtin_amdgcn_wmma_f32_16x16x32_bf16(false, pf.v, false, vf.v, (short)0, accO[j], false, false);
    }
  }

  // normalize and store to attn buffer [s][u*128 + d] fp32
#pragma unroll
  for (int r = 0; r < 8; ++r) {
    float inv = 1.f / lrow[r];
    float* orow = ob + (size_t)(qt + r + 8 * half) * 4096 + u * 128;
#pragma unroll
    for (int j = 0; j < 8; ++j) orow[j * 16 + m] = accO[j][r] * inv;
  }
}

// ---------------------------------------------------------------------------
// Launch: projections (WMMA GEMM) -> RoPE/convert -> SWA attention -> O-proj
// ---------------------------------------------------------------------------
extern "C" void kernel_launch(void* const* d_in, const int* in_sizes, int n_in,
                              void* d_out, int out_size, void* d_ws, size_t ws_size,
                              hipStream_t stream)
{
  const float* hs = (const float*)d_in[0];
  // d_in[1] = attention_mask (sliding-window causal, recomputed analytically)
  const float* Wq = (const float*)d_in[2];
  const float* Wk = (const float*)d_in[3];
  const float* Wv = (const float*)d_in[4];
  const float* Wo = (const float*)d_in[5];
  float* out = (float*)d_out;

  char* ws = (char*)d_ws;
  float* q_pre = (float*)ws;                 ws += (size_t)2048 * 4096 * 4;
  float* k_pre = (float*)ws;                 ws += (size_t)2048 * 1024 * 4;
  float* v_pre = (float*)ws;                 ws += (size_t)2048 * 1024 * 4;
  float* attnb = (float*)ws;                 ws += (size_t)2048 * 4096 * 4;
  unsigned short* qbuf = (unsigned short*)ws; ws += (size_t)2048 * 4096 * 2;
  unsigned short* kbuf = (unsigned short*)ws; ws += (size_t)2048 * 1024 * 2;
  unsigned short* vbuf = (unsigned short*)ws; ws += (size_t)2048 * 1024 * 2;

  dim3 blk(256);
  gemm_f32_bf16wmma<<<dim3(4096 / BN, 2048 / BM), blk, 0, stream>>>(hs, Wq, q_pre, 2048, 4096, 4096);
  gemm_f32_bf16wmma<<<dim3(1024 / BN, 2048 / BM), blk, 0, stream>>>(hs, Wk, k_pre, 2048, 1024, 4096);
  gemm_f32_bf16wmma<<<dim3(1024 / BN, 2048 / BM), blk, 0, stream>>>(hs, Wv, v_pre, 2048, 1024, 4096);

  rope_transpose_bf16<<<(2048 * 32 * 64) / 256, blk, 0, stream>>>(q_pre, qbuf, 32);
  rope_transpose_bf16<<<(2048 * 8 * 64) / 256, blk, 0, stream>>>(k_pre, kbuf, 8);
  conv_transpose_bf16<<<(2048 * 8 * 128) / 256, blk, 0, stream>>>(v_pre, vbuf, 8);

  attn_swa_kernel<<<dim3(32, 32), dim3(128), 0, stream>>>(qbuf, kbuf, vbuf, attnb);

  gemm_f32_bf16wmma<<<dim3(4096 / BN, 2048 / BM), blk, 0, stream>>>(attnb, Wo, out, 2048, 4096, 4096);
}
